// Model_66383014527431
// MI455X (gfx1250) — compile-verified, matrix operand order checked by reading, not run on hardware
//
#include <hip/hip_runtime.h>
#include <hip/hip_bf16.h>

typedef __attribute__((ext_vector_type(16))) _Float16 v16h;
typedef __attribute__((ext_vector_type(8)))  _Float16 v8h;
typedef __attribute__((ext_vector_type(8)))  float    v8f;

#define NUM_STATES 32768

// ---------------------------------------------------------------------------
// Hardware tanh: CDNA5 has V_TANH_F32 (TRANS32 op).  Use the gfx1250 builtin
// when the toolchain declares it, else emit the instruction directly.
// ---------------------------------------------------------------------------
__device__ __forceinline__ float fast_tanh(float x) {
#if __has_builtin(__builtin_amdgcn_tanhf)
    return __builtin_amdgcn_tanhf(x);
#else
    float r;
    asm volatile("v_tanh_f32 %0, %1" : "=v"(r) : "v"(x));
    return r;
#endif
}

// ---------------------------------------------------------------------------
// Fragment loaders.  CDNA5 16-bit A/B WMMA layout (wave32):
//   lanes 0-15  : halves 0..7 -> K = base+0..7,   halves 8..15 -> K = base+16..23
//   lanes 16-31 : halves 0..7 -> K = base+8..15,  halves 8..15 -> K = base+24..31
// i.e. per lane: two contiguous 8-half (16-byte) chunks at +hiOff and +hiOff+16,
// where hiOff = (lane>=16) ? 8 : 0.  A: lane%16 = M row.  B: lane%16 = N col
// (we store weights transposed [N][K] so B rows are contiguous in K).
// ---------------------------------------------------------------------------
__device__ __forceinline__ v16h load_frag_h(const _Float16* p, int hiOff) {
    v16h f;
    ((v8h*)&f)[0] = *(const v8h*)(p + hiOff);
    ((v8h*)&f)[1] = *(const v8h*)(p + hiOff + 16);
    return f;
}

__device__ __forceinline__ v16h load_frag_f32(const float* p, int hiOff) {
    v16h f;
#pragma unroll
    for (int i = 0; i < 8; ++i) {
        f[i]     = (_Float16)p[hiOff + i];
        f[8 + i] = (_Float16)p[hiOff + 16 + i];
    }
    return f;
}

__device__ __forceinline__ v8f wmma_f16(v16h a, v16h b, v8f c) {
    return __builtin_amdgcn_wmma_f32_16x16x32_f16(
        /*neg_a=*/false, a, /*neg_b=*/false, b,
        /*c_mod=*/(short)0, c, /*reuse_a=*/false, /*reuse_b=*/false);
}

// ---------------------------------------------------------------------------
// Kernel 0: fp32 -> f16 weight transpose+convert.
//   src: row-major [K][N] fp32.  dst: row-major [Npad][K] f16, zero padded.
// ---------------------------------------------------------------------------
__global__ void cvt_transpose_f16(const float* __restrict__ src,
                                  _Float16* __restrict__ dst,
                                  int K, int N, int Npad) {
    int idx = blockIdx.x * blockDim.x + threadIdx.x;
    int total = K * Npad;
    if (idx >= total) return;
    int n = idx / K;
    int k = idx - n * K;
    dst[idx] = (n < N) ? (_Float16)src[(size_t)k * N + n] : (_Float16)0.0f;
}

// ---------------------------------------------------------------------------
// Kernel 1: fused ball MLP (32->512->512->256, tanh) + per-state max pool.
// One wave per 16-ball tile == 2 states.  128 threads = 4 waves per block.
// Per-wave 16KB LDS staging buffer for the C-layout -> A-layout transpose.
// ---------------------------------------------------------------------------
__global__ __launch_bounds__(128) void ball_mlp_kernel(
    const float* __restrict__ balls,
    const float* __restrict__ b1, const float* __restrict__ b2,
    const float* __restrict__ b3,
    const _Float16* __restrict__ W1t,   // [512][32]
    const _Float16* __restrict__ W2t,   // [512][512]
    const _Float16* __restrict__ W3t,   // [256][512]
    _Float16* __restrict__ vh)          // [32768][256] pooled output (f16)
{
    __shared__ _Float16 smem[4 * 16 * 512];   // 64 KB: 16KB per wave

    const int lane   = threadIdx.x & 31;
    const int lane15 = lane & 15;
    const int hiOff  = (lane >> 4) << 3;      // 0 or 8 (K chunk offset)
    const int moff   = (lane >> 4) << 3;      // C-layout row offset
    const int wave   = threadIdx.x >> 5;
    _Float16* h = smem + wave * (16 * 512);

    const int tile     = blockIdx.x * 4 + wave;   // 0..16383
    const int ballBase = tile * 16;

    // ---- layer 1: x[16x32] @ W1 -> h1[16x512] ----
    v16h a0 = load_frag_f32(balls + (size_t)(ballBase + lane15) * 32, hiOff);
    for (int nc = 0; nc < 32; ++nc) {
        const int col = nc * 16 + lane15;
        v16h bf = load_frag_h(W1t + (size_t)col * 32, hiOff);
        float bias = b1[col];
        v8f acc;
#pragma unroll
        for (int i = 0; i < 8; ++i) acc[i] = bias;
        acc = wmma_f16(a0, bf, acc);
#pragma unroll
        for (int r = 0; r < 8; ++r)
            h[(size_t)(r + moff) * 512 + col] = (_Float16)fast_tanh(acc[r]);
    }

    // pull full K=512 A-fragment set into registers (16 x v16h)
    v16h af[16];
#pragma unroll
    for (int k = 0; k < 16; ++k)
        af[k] = load_frag_h(h + (size_t)lane15 * 512 + k * 32, hiOff);

    // ---- layer 2: h1 @ W2 -> h2[16x512] ----
    for (int nc = 0; nc < 32; ++nc) {
        const int col = nc * 16 + lane15;
        if (nc + 1 < 32)
            __builtin_prefetch(W2t + (size_t)(col + 16) * 512, 0, 1);
        float bias = b2[col];
        v8f acc;
#pragma unroll
        for (int i = 0; i < 8; ++i) acc[i] = bias;
#pragma unroll
        for (int k = 0; k < 16; ++k) {
            v16h bf = load_frag_h(W2t + (size_t)col * 512 + k * 32, hiOff);
            acc = wmma_f16(af[k], bf, acc);
        }
#pragma unroll
        for (int r = 0; r < 8; ++r)
            h[(size_t)(r + moff) * 512 + col] = (_Float16)fast_tanh(acc[r]);
    }

#pragma unroll
    for (int k = 0; k < 16; ++k)
        af[k] = load_frag_h(h + (size_t)lane15 * 512 + k * 32, hiOff);

    // ---- layer 3: h2 @ W3 -> tanh -> segment max over 8 balls ----
    // C layout: lanes 0-15 hold rows M=0..7 (state 2*tile),
    //           lanes 16-31 hold rows M=8..15 (state 2*tile+1).
    // tanh is monotonic: max(tanh(x)) == tanh(max(x)) -> reduce first, 1 tanh.
    const int s = tile * 2 + (lane >> 4);
    for (int nc = 0; nc < 16; ++nc) {
        const int col = nc * 16 + lane15;
        float bias = b3[col];
        v8f acc;
#pragma unroll
        for (int i = 0; i < 8; ++i) acc[i] = bias;
#pragma unroll
        for (int k = 0; k < 16; ++k) {
            v16h bf = load_frag_h(W3t + (size_t)col * 512 + k * 32, hiOff);
            acc = wmma_f16(af[k], bf, acc);
        }
        float m = acc[0];
#pragma unroll
        for (int r = 1; r < 8; ++r) m = fmaxf(m, acc[r]);
        vh[(size_t)s * 256 + col] = (_Float16)fast_tanh(m);
    }
}

// ---------------------------------------------------------------------------
// Kernel 2: heads.  v[32768x256] @ {Ws,Wc,Wp,Wk} + bias -> fp32 outputs.
// grid.x: 512 blocks x 4 waves -> 2048 row tiles of 16 states.
// grid.y: 111 column tiles = 23(shoot) + 23(scratch) + 64(place) + 1(pick).
// ---------------------------------------------------------------------------
__global__ __launch_bounds__(128) void pool_heads_kernel(
    const _Float16* __restrict__ vh,
    const _Float16* __restrict__ Wst, const float* __restrict__ bs,
    const _Float16* __restrict__ Wct, const float* __restrict__ bc,
    const _Float16* __restrict__ Wpt, const float* __restrict__ bp,
    const _Float16* __restrict__ Wkt, const float* __restrict__ bk,
    float* __restrict__ out)
{
    const int lane   = threadIdx.x & 31;
    const int lane15 = lane & 15;
    const int hiOff  = (lane >> 4) << 3;
    const int wave   = threadIdx.x >> 5;
    const int rowTile = blockIdx.x * 4 + wave;     // 0..2047
    const int y = blockIdx.y;                      // 0..110

    const _Float16* W; const float* b; int width; int nc; size_t outOff;
    if (y < 23)       { W = Wst; b = bs; width = 360;  nc = y;      outOff = 0; }
    else if (y < 46)  { W = Wct; b = bc; width = 360;  nc = y - 23; outOff = (size_t)NUM_STATES * 360; }
    else if (y < 110) { W = Wpt; b = bp; width = 1024; nc = y - 46; outOff = (size_t)NUM_STATES * 720; }
    else              { W = Wkt; b = bk; width = 6;    nc = 0;      outOff = (size_t)NUM_STATES * 1744; }

    // A fragments: 16 states x K=256 from pooled vectors
    v16h af[8];
#pragma unroll
    for (int k = 0; k < 8; ++k)
        af[k] = load_frag_h(vh + (size_t)(rowTile * 16 + lane15) * 256 + k * 32, hiOff);

    const int col = nc * 16 + lane15;              // < padded width, always valid in W
    float bias = (col < width) ? b[col] : 0.0f;
    v8f acc;
#pragma unroll
    for (int i = 0; i < 8; ++i) acc[i] = bias;

#pragma unroll
    for (int k = 0; k < 8; ++k) {
        v16h bf = load_frag_h(W + (size_t)col * 256 + k * 32, hiOff);
        acc = wmma_f16(af[k], bf, acc);
    }

    if (col < width) {
        const int rowBase = rowTile * 16 + ((lane >> 4) << 3);
#pragma unroll
        for (int r = 0; r < 8; ++r)
            out[outOff + (size_t)(rowBase + r) * width + col] = acc[r];
    }
}

// ---------------------------------------------------------------------------
// Host launch
// ---------------------------------------------------------------------------
extern "C" void kernel_launch(void* const* d_in, const int* in_sizes, int n_in,
                              void* d_out, int out_size, void* d_ws, size_t ws_size,
                              hipStream_t stream) {
    const float* balls = (const float*)d_in[0];
    // d_in[1]=segment_ids, d_in[2]=num_segments : implicit in contiguous layout
    const float* W1 = (const float*)d_in[3];
    const float* b1 = (const float*)d_in[4];
    const float* W2 = (const float*)d_in[5];
    const float* b2 = (const float*)d_in[6];
    const float* W3 = (const float*)d_in[7];
    const float* b3 = (const float*)d_in[8];
    const float* Ws = (const float*)d_in[9];
    const float* bs = (const float*)d_in[10];
    const float* Wc = (const float*)d_in[11];
    const float* bc = (const float*)d_in[12];
    const float* Wp = (const float*)d_in[13];
    const float* bp = (const float*)d_in[14];
    const float* Wk = (const float*)d_in[15];
    const float* bk = (const float*)d_in[16];

    _Float16* wsh = (_Float16*)d_ws;
    _Float16* W1t = wsh;                      // [512][32]
    _Float16* W2t = W1t + 512 * 32;           // [512][512]
    _Float16* W3t = W2t + 512 * 512;          // [256][512]
    _Float16* Wst = W3t + 256 * 512;          // [368][256] (360 padded)
    _Float16* Wct = Wst + 368 * 256;          // [368][256]
    _Float16* Wpt = Wct + 368 * 256;          // [1024][256]
    _Float16* Wkt = Wpt + 1024 * 256;         // [16][256]  (6 padded)
    _Float16* vh  = Wkt + 16 * 256;           // [32768][256]

    auto cvt = [&](const float* src, _Float16* dst, int K, int N, int Npad) {
        int total = K * Npad;
        cvt_transpose_f16<<<(total + 255) / 256, 256, 0, stream>>>(src, dst, K, N, Npad);
    };
    cvt(W1, W1t, 32, 512, 512);
    cvt(W2, W2t, 512, 512, 512);
    cvt(W3, W3t, 512, 256, 256);
    cvt(Ws, Wst, 256, 360, 368);
    cvt(Wc, Wct, 256, 360, 368);
    cvt(Wp, Wpt, 256, 1024, 1024);
    cvt(Wk, Wkt, 256, 6, 16);

    // 16384 ball tiles / 4 waves per block
    ball_mlp_kernel<<<4096, 128, 0, stream>>>(balls, b1, b2, b3, W1t, W2t, W3t, vh);

    // 2048 row tiles / 4 waves per block; 111 column tiles across 4 heads
    pool_heads_kernel<<<dim3(512, 111), 128, 0, stream>>>(
        vh, Wst, bs, Wct, bc, Wpt, bp, Wkt, bk, (float*)d_out);
}
